// HeteroRGCNLayer_3822520893714
// MI455X (gfx1250) — matrix operand order repaired and checked by reading, not compiled, not run on hardware
//
#include <hip/hip_runtime.h>
#include <hip/hip_bf16.h>

typedef __attribute__((ext_vector_type(16))) __bf16 v16bf;
typedef __attribute__((ext_vector_type(8)))  float  v8f;

constexpr int N_NODES  = 50000;
constexpr int IN_SIZE  = 256;
constexpr int OUT_SIZE = 128;
constexpr int N_ETYPES = 3;
constexpr int N_EDGES  = 1000000;
constexpr int M_TILES  = N_NODES / 16;   // 3125 (exact)
constexpr int K_STEPS  = IN_SIZE / 32;   // 8
constexpr int N_TILES  = OUT_SIZE / 16;  // 8
// packed B fragments: [e][ks][t][lane][16] bf16 (one v16bf per lane per tile)
constexpr int B_FRAGS  = N_ETYPES * K_STEPS * N_TILES * 32;   // 6144 fragments

// ---------------------------------------------------------------------------
// Zero-fill helper (vectorized, grid-stride). n4 = element count / 4.
// ---------------------------------------------------------------------------
__global__ __launch_bounds__(256) void zero_kernel(float* __restrict__ p, size_t n4) {
    size_t i = (size_t)blockIdx.x * blockDim.x + threadIdx.x;
    size_t stride = (size_t)gridDim.x * blockDim.x;
    float4 z = make_float4(0.f, 0.f, 0.f, 0.f);
    for (; i < n4; i += stride) ((float4*)p)[i] = z;
}

// ---------------------------------------------------------------------------
// Stage P: pre-pack W into split-bf16 (hi+lo) B fragments in the exact
// wave32 WMMA B layout (ISA 7.12.2): lane group g holds K 8g..8g+7, 16+8g..16+8g+7,
// N = lane&15. One thread per (e, kstep, ntile, lane) -> one v16bf hi + lo.
// Each W element is converted exactly once for the whole launch.
// ---------------------------------------------------------------------------
__global__ __launch_bounds__(256) void pack_w_kernel(
    const float* __restrict__ W,      // [N_ETYPES, IN_SIZE, OUT_SIZE]
    __bf16* __restrict__ Bhi,         // [B_FRAGS * 16]
    __bf16* __restrict__ Blo)         // [B_FRAGS * 16]
{
    const int tid = blockIdx.x * blockDim.x + threadIdx.x;
    if (tid >= B_FRAGS) return;
    const int lane = tid & 31;
    const int t    = (tid >> 5) & 7;
    const int ks   = (tid >> 8) & 7;
    const int e    = tid >> 11;
    const int g    = lane >> 4;
    const int ncol = lane & 15;
    const int bbase = g * 16;

    const float* __restrict__ We = W + (size_t)e * IN_SIZE * OUT_SIZE;
    v16bf hi, lo;
#pragma unroll
    for (int i = 0; i < 16; ++i) {
        const float v = We[(size_t)(ks * 32 + (bbase >> 1) + (i & 7) + (i >> 3) * 16) * OUT_SIZE + t * 16 + ncol];
        const __bf16 h = (__bf16)v;
        hi[i] = h;
        lo[i] = (__bf16)(v - (float)h);
    }
    ((v16bf*)Bhi)[tid] = hi;
    ((v16bf*)Blo)[tid] = lo;
}

// ---------------------------------------------------------------------------
// Stage 1: Wh[e] = x @ W[e] + b[e] via v_wmma_f32_16x16x32_bf16 with
// split-bf16 (A_hi*B_hi + A_hi*B_lo + A_lo*B_hi) ~ f32 accuracy.
// One wave computes a 16x128 strip; B fragments stream straight from the
// packed planes as 32B vector loads (no conversion in the hot loop).
// ---------------------------------------------------------------------------
__global__ __launch_bounds__(256) void rgcn_gemm_wmma(
    const float*  __restrict__ x,     // [N_NODES, IN_SIZE]
    const __bf16* __restrict__ Bhi,   // packed B hi plane
    const __bf16* __restrict__ Blo,   // packed B lo plane
    const float*  __restrict__ bias,  // [N_ETYPES, OUT_SIZE]
    float* __restrict__ Wh)           // [N_ETYPES, N_NODES, OUT_SIZE]
{
    const int e     = blockIdx.y;
    const int wave  = threadIdx.x >> 5;
    const int lane  = threadIdx.x & 31;
    const int mtile = blockIdx.x * 8 + wave;
    if (mtile >= M_TILES) return;                 // wave-uniform branch

    const int row0  = mtile * 16;
    const int g     = lane >> 4;
    const int ncol  = lane & 15;
    const int abase = g * 8;                      // A: grp1 holds K+8 / K+24 halves

    const float* __restrict__ xrow = x + (size_t)(row0 + ncol) * IN_SIZE;
    const v16bf* __restrict__ BhiV = (const v16bf*)Bhi;
    const v16bf* __restrict__ BloV = (const v16bf*)Blo;

    v8f acc[N_TILES] = {};

    for (int ks = 0; ks < K_STEPS; ++ks) {
        const int K0 = ks * 32;
        if (ks + 1 < K_STEPS)                     // gfx1250 global_prefetch_b8
            __builtin_prefetch(xrow + K0 + 32, 0, 3);

        // ---- A fragment: two contiguous 8-float segments -> 4x b128 ----
        const float4 a0 = *(const float4*)(xrow + K0 + abase);
        const float4 a1 = *(const float4*)(xrow + K0 + abase + 4);
        const float4 a2 = *(const float4*)(xrow + K0 + abase + 16);
        const float4 a3 = *(const float4*)(xrow + K0 + abase + 20);
        const float af[16] = { a0.x, a0.y, a0.z, a0.w, a1.x, a1.y, a1.z, a1.w,
                               a2.x, a2.y, a2.z, a2.w, a3.x, a3.y, a3.z, a3.w };
        v16bf a_hi, a_lo;
#pragma unroll
        for (int i = 0; i < 16; ++i) {
            const __bf16 h = (__bf16)af[i];
            a_hi[i] = h;
            a_lo[i] = (__bf16)(af[i] - (float)h);
        }

        // ---- 8 B tiles: pure 32B vector loads from packed planes ----
        const size_t fragbase = ((size_t)(e * K_STEPS + ks) * N_TILES) * 32 + lane;
#pragma unroll
        for (int t = 0; t < N_TILES; ++t) {
            const v16bf b_hi = BhiV[fragbase + (size_t)t * 32];
            const v16bf b_lo = BloV[fragbase + (size_t)t * 32];
            acc[t] = __builtin_amdgcn_wmma_f32_16x16x32_bf16(
                false, a_hi, false, b_hi, (short)0, acc[t], false, false);
            acc[t] = __builtin_amdgcn_wmma_f32_16x16x32_bf16(
                false, a_hi, false, b_lo, (short)0, acc[t], false, false);
            acc[t] = __builtin_amdgcn_wmma_f32_16x16x32_bf16(
                false, a_lo, false, b_hi, (short)0, acc[t], false, false);
        }
    }

    // ---- bias + store. C/D layout: VGPR r -> row r + 8*g, col = lane&15 ----
    const float* __restrict__ be = bias + (size_t)e * OUT_SIZE;
    const int rbase = row0 + g * 8;
#pragma unroll
    for (int t = 0; t < N_TILES; ++t) {
        const int col = t * 16 + ncol;
        const float bv = be[col];
#pragma unroll
        for (int r = 0; r < 8; ++r) {
            Wh[((size_t)e * N_NODES + rbase + r) * OUT_SIZE + col] = acc[t][r] + bv;
        }
    }
}

// ---------------------------------------------------------------------------
// Stage 2: edge-parallel gather + scatter-add. One wave per edge; each lane
// moves 4 contiguous floats. Wh (76.8 MB) and the accumulator (76.8 MB) both
// fit in the 192 MB L2, so gathers and f32 atomics stay on-die.
// ---------------------------------------------------------------------------
__global__ __launch_bounds__(256) void rgcn_scatter(
    const float* __restrict__ Wh,
    const int* __restrict__ esrc, const int* __restrict__ edst,
    float* __restrict__ out, float* __restrict__ deg)
{
    const long long idx = (long long)blockIdx.x * 8 + (threadIdx.x >> 5);
    const int lane = threadIdx.x & 31;
    if (idx >= (long long)N_ETYPES * N_EDGES) return;

    const int e   = (int)(idx / N_EDGES);
    const int eid = (int)(idx % N_EDGES);
    const int src = esrc[(size_t)e * N_EDGES + eid];
    const int dst = edst[(size_t)e * N_EDGES + eid];

    const float4 v = ((const float4*)(Wh + ((size_t)e * N_NODES + src) * OUT_SIZE))[lane];
    float* op = out + ((size_t)dst * N_ETYPES + e) * OUT_SIZE + lane * 4;
    atomicAdd(op + 0, v.x);
    atomicAdd(op + 1, v.y);
    atomicAdd(op + 2, v.z);
    atomicAdd(op + 3, v.w);
    if (lane == 0) atomicAdd(&deg[(size_t)e * N_NODES + dst], 1.0f);
}

// ---------------------------------------------------------------------------
// Stage 3: mean-normalize: out = deg > 0 ? sum / deg : 0.  float4 per thread.
// ---------------------------------------------------------------------------
__global__ __launch_bounds__(256) void rgcn_normalize(
    float* __restrict__ out, const float* __restrict__ deg)
{
    const size_t i = (size_t)blockIdx.x * blockDim.x + threadIdx.x;
    const size_t total4 = (size_t)N_NODES * N_ETYPES * OUT_SIZE / 4;
    if (i >= total4) return;
    const size_t base = i * 4;
    const int n = (int)(base / (N_ETYPES * OUT_SIZE));
    const int e = (int)((base % (N_ETYPES * OUT_SIZE)) / OUT_SIZE);
    const float d = deg[(size_t)e * N_NODES + n];
    const float s = d > 0.f ? 1.f / d : 0.f;
    float4 v = ((float4*)out)[i];
    v.x *= s; v.y *= s; v.z *= s; v.w *= s;
    ((float4*)out)[i] = v;
}

// ---------------------------------------------------------------------------
extern "C" void kernel_launch(void* const* d_in, const int* in_sizes, int n_in,
                              void* d_out, int out_size, void* d_ws, size_t ws_size,
                              hipStream_t stream) {
    const float* x    = (const float*)d_in[0];
    const int*   esrc = (const int*)  d_in[1];
    const int*   edst = (const int*)  d_in[2];
    const float* W    = (const float*)d_in[3];
    const float* b    = (const float*)d_in[4];
    float* out = (float*)d_out;

    // Workspace layout (floats): Wh | Bhi | Blo | deg   (all 32B aligned)
    const size_t wh_elems  = (size_t)N_ETYPES * N_NODES * OUT_SIZE;  // 19.2M
    const size_t bpk_floats = (size_t)B_FRAGS * 16 / 2;              // bf16 plane in floats
    float* Wh   = (float*)d_ws;
    __bf16* Bhi = (__bf16*)(Wh + wh_elems);
    __bf16* Blo = (__bf16*)(Wh + wh_elems + bpk_floats);
    float* deg  = Wh + wh_elems + 2 * bpk_floats;

    const size_t out_elems = wh_elems;                    // [N,3,128]
    const size_t deg_elems = (size_t)N_ETYPES * N_NODES;  // 150K

    // Stage 0: zero accumulator and degree arrays.
    zero_kernel<<<2048, 256, 0, stream>>>(out, out_elems / 4);
    zero_kernel<<<256, 256, 0, stream>>>(deg, deg_elems / 4);

    // Stage P: one-time split-bf16 pre-pack of W into B-fragment layout.
    pack_w_kernel<<<(B_FRAGS + 255) / 256, 256, 0, stream>>>(W, Bhi, Blo);

    // Stage 1: WMMA GEMM. 3125 M-tiles, 8 waves/block; grid.y = etype.
    dim3 ggrid((M_TILES + 7) / 8, N_ETYPES, 1);
    rgcn_gemm_wmma<<<ggrid, 256, 0, stream>>>(x, Bhi, Blo, b, Wh);

    // Stage 2: edge scatter. 3M edges, 8 edges (waves) per block.
    const long long n_edges_total = (long long)N_ETYPES * N_EDGES;
    rgcn_scatter<<<(unsigned)((n_edges_total + 7) / 8), 256, 0, stream>>>(
        Wh, esrc, edst, out, deg);

    // Stage 3: normalize to mean.
    rgcn_normalize<<<(unsigned)((out_elems / 4 + 255) / 256), 256, 0, stream>>>(out, deg);
}
